// Transformer_1967095021882
// MI455X (gfx1250) — matrix operand order repaired and checked by reading, not compiled, not run on hardware
//
#include <hip/hip_runtime.h>

// ---------------------------------------------------------------------------
// CDNA5 / gfx1250 point-transformer block.
// Matmuls: v_wmma_f32_16x16x32_f16 (f16 operands, f32 accum), wave32.
// Fragment K order: contiguous k = 16*half + j, same in A and B.
// Async global->LDS staging (ASYNCcnt) used when the builtin is available.
// ---------------------------------------------------------------------------

typedef __attribute__((ext_vector_type(16))) _Float16 v16h;
typedef __attribute__((ext_vector_type(8)))  _Float16 v8h;
typedef __attribute__((ext_vector_type(8)))  float    v8f;
typedef __attribute__((ext_vector_type(4)))  int      v4i;

static constexpr int B_   = 2;
static constexpr int CIN  = 128;
static constexpr int N_   = 2048;
static constexpr int DIM  = 256;
static constexpr int MANC = 128;          // anchors
static constexpr int KNN  = 16;
static constexpr int HA   = 1024;         // attn hidden
static constexpr int HP   = 64;           // pos/lgrp hidden
static constexpr int NPIX = N_ * KNN;     // 32768 pixels per batch

#ifndef __has_builtin
#define __has_builtin(x) 0
#endif
#if __has_builtin(__builtin_amdgcn_global_load_async_to_lds_b128)
#define USE_ASYNC_LDS 1
// builtin signature (from compiler diagnostic): arg0 = AS1 (global) v4i*,
// arg1 = AS3 (LDS) v4i*, then imm offset, imm cpol.
typedef __attribute__((address_space(1))) v4i* gas_v4i;
typedef __attribute__((address_space(3))) v4i* las_v4i;
#else
#define USE_ASYNC_LDS 0
#endif

__device__ __forceinline__ void wait_async_zero() {
#if __has_builtin(__builtin_amdgcn_s_wait_asynccnt)
  __builtin_amdgcn_s_wait_asynccnt(0);
#else
  asm volatile("s_wait_asynccnt 0" ::: "memory");
#endif
}

__device__ __forceinline__ v8f wmma32f16(v16h a, v16h b, v8f c) {
  // D = A(16x32 f16) * B(32x16 f16) + C(16x16 f32)
  return __builtin_amdgcn_wmma_f32_16x16x32_f16(false, a, false, b, (short)0, c,
                                                false, false);
}

// A fragment (16x32): lane row = lane&15, half = lane>>4.
// elem j -> K = k0 + half*16 + j.
// K runs are contiguous -> vector loads.
__device__ __forceinline__ v16h frag_a_f32(const float* __restrict__ W, int row,
                                           int ld, int k0, int half) {
  const float* p = W + (size_t)row * ld + k0 + half * 16;
  float4 a0 = *(const float4*)(p);
  float4 a1 = *(const float4*)(p + 4);
  float4 b0 = *(const float4*)(p + 8);
  float4 b1 = *(const float4*)(p + 12);
  v16h a;
  a[0] = (_Float16)a0.x;  a[1] = (_Float16)a0.y;
  a[2] = (_Float16)a0.z;  a[3] = (_Float16)a0.w;
  a[4] = (_Float16)a1.x;  a[5] = (_Float16)a1.y;
  a[6] = (_Float16)a1.z;  a[7] = (_Float16)a1.w;
  a[8] = (_Float16)b0.x;  a[9] = (_Float16)b0.y;
  a[10] = (_Float16)b0.z; a[11] = (_Float16)b0.w;
  a[12] = (_Float16)b1.x; a[13] = (_Float16)b1.y;
  a[14] = (_Float16)b1.z; a[15] = (_Float16)b1.w;
  return a;
}

__device__ __forceinline__ v16h frag_a_f16(const _Float16* __restrict__ W, int row,
                                           int ld, int k0, int half) {
  const _Float16* p = W + (size_t)row * ld + k0 + half * 16;
  v8h lo = *(const v8h*)(p);
  v8h hi = *(const v8h*)(p + 8);
  return __builtin_shufflevector(lo, hi, 0, 1, 2, 3, 4, 5, 6, 7, 8, 9, 10, 11,
                                 12, 13, 14, 15);
}

// B fragment (32x16) from row-major KxN: col = lane&15, elem j -> K=k0+half*16+j.
__device__ __forceinline__ v16h frag_b_f32(const float* __restrict__ X, int k0,
                                           size_t ld, int col, int half) {
  v16h b;
  const float* p = X + (size_t)(k0 + half * 16) * ld + col;
#pragma unroll
  for (int j = 0; j < 16; ++j) b[j] = (_Float16)p[(size_t)j * ld];
  return b;
}

__device__ __forceinline__ v16h frag_b_f16(const _Float16* __restrict__ X, int k0,
                                           size_t ld, int col, int half) {
  v16h b;
  const _Float16* p = X + (size_t)(k0 + half * 16) * ld + col;
#pragma unroll
  for (int j = 0; j < 16; ++j) b[j] = p[(size_t)j * ld];
  return b;
}

// 16-lane (half-wave) reductions; masks <16 stay inside the half on wave32.
__device__ __forceinline__ float redmax16(float x) {
#pragma unroll
  for (int m = 8; m >= 1; m >>= 1) x = fmaxf(x, __shfl_xor(x, m, 32));
  return x;
}
__device__ __forceinline__ float redsum16(float x) {
#pragma unroll
  for (int m = 8; m >= 1; m >>= 1) x += __shfl_xor(x, m, 32);
  return x;
}

// ---------------------------------------------------------------------------
__global__ void cvt_f32_to_f16(const float* __restrict__ src,
                               _Float16* __restrict__ dst, int nElem) {
  for (int i = blockIdx.x * blockDim.x + threadIdx.x; i < nElem;
       i += gridDim.x * blockDim.x)
    dst[i] = (_Float16)src[i];
}

// ---------------------------------------------------------------------------
// 16-NN brute force, per-thread sorted insertion. (Both KNNs in the reference
// are over the same point set -> one pass.)
__global__ void knn_kernel(const float* __restrict__ pos, int* __restrict__ idx) {
  int t = blockIdx.x * blockDim.x + threadIdx.x;
  if (t >= B_ * N_) return;
  int b = t / N_, q = t - b * N_;
  const float* px = pos + (size_t)b * 3 * N_;
  float qx = px[q], qy = px[N_ + q], qz = px[2 * N_ + q];
  float bd[KNN];
  int bi[KNN];
#pragma unroll
  for (int i = 0; i < KNN; ++i) { bd[i] = 3.4e38f; bi[i] = 0; }
  for (int c = 0; c < N_; ++c) {
    float dx = qx - px[c], dy = qy - px[N_ + c], dz = qz - px[2 * N_ + c];
    float d = dx * dx + dy * dy + dz * dz;
    if (d < bd[KNN - 1]) {
      bd[KNN - 1] = d; bi[KNN - 1] = c;
#pragma unroll
      for (int i = KNN - 1; i >= 1; --i) {
        if (bd[i] < bd[i - 1]) {
          float td = bd[i]; bd[i] = bd[i - 1]; bd[i - 1] = td;
          int ti = bi[i]; bi[i] = bi[i - 1]; bi[i - 1] = ti;
        }
      }
    }
  }
#pragma unroll
  for (int i = 0; i < KNN; ++i) idx[(size_t)t * KNN + i] = bi[i];
}

// ---------------------------------------------------------------------------
// 1x1-conv GEMM: Y[b] = W(OxC) @ X[b](CxN) + bias (+ resid).
// 256 threads = 8 independent waves; each wave computes a 16x64 strip
// (A fragment reused across 4 WMMAs per K-step).
template <bool HAS_RES>
__global__ __launch_bounds__(256) void gemm_wave(
    const float* __restrict__ W, const float* __restrict__ X,
    const float* __restrict__ bias, const float* __restrict__ resid,
    float* __restrict__ Y, int O, int C, int N) {
  int lane = threadIdx.x & 31;
  int wave = threadIdx.x >> 5;
  int half = lane >> 4, row = lane & 15;
  int n0 = blockIdx.x * 64;
  int o0 = (blockIdx.y * 8 + wave) * 16;
  int b = blockIdx.z;
  const float* Xb = X + (size_t)b * C * N;
  v8f acc[4] = {{}, {}, {}, {}};
  for (int k0 = 0; k0 < C; k0 += 32) {
    v16h a = frag_a_f32(W, o0 + row, C, k0, half);
#pragma unroll
    for (int t = 0; t < 4; ++t)
      acc[t] = wmma32f16(a, frag_b_f32(Xb, k0, (size_t)N, n0 + t * 16 + row, half),
                         acc[t]);
  }
  size_t ybase = (size_t)b * O * N;
#pragma unroll
  for (int t = 0; t < 4; ++t) {
#pragma unroll
    for (int v = 0; v < 8; ++v) {
      int oo = o0 + v + 8 * half;
      float yv = acc[t][v] + bias[oo];
      size_t off = ybase + (size_t)oo * N + n0 + t * 16 + row;
      if constexpr (HAS_RES) yv += resid[off];
      Y[off] = yv;
    }
  }
}

// ---------------------------------------------------------------------------
// Per-(b,q,j) pixel: anchor geometry + tiny MLPs (lgrp 6->32->64, pos
// 3->64 conv+bn+relu). Writes h64l/h64p f16 in [b][ch][pixel] layout
// (K-major for the following WMMA B fragments).
__global__ __launch_bounds__(256) void prep_kernel(
    const float* __restrict__ pos, const int* __restrict__ idx,
    const int* __restrict__ map_idx, const float* __restrict__ P_anchor,
    const float* __restrict__ anchor_dist, const float* __restrict__ lw1,
    const float* __restrict__ lw2, const float* __restrict__ pw1,
    const float* __restrict__ pb1, const float* __restrict__ pg1,
    const float* __restrict__ pbt1, _Float16* __restrict__ h64l,
    _Float16* __restrict__ h64p) {
  __shared__ float s_lw1[32 * 6];
  __shared__ float s_lw2[64 * 32];
  __shared__ float s_pw1[64 * 3];
  __shared__ float s_pm[64 * 3];  // [bias | bn-scale | bn-beta]
  for (int i = threadIdx.x; i < 32 * 6; i += 256) s_lw1[i] = lw1[i];
  for (int i = threadIdx.x; i < 64 * 32; i += 256) s_lw2[i] = lw2[i];
  for (int i = threadIdx.x; i < 64 * 3; i += 256) s_pw1[i] = pw1[i];
  for (int i = threadIdx.x; i < 64; i += 256) {
    s_pm[i]       = pb1[i];
    s_pm[64 + i]  = pg1[i] * rsqrtf(1.0f + 1e-5f);
    s_pm[128 + i] = pbt1[i];
  }
  __syncthreads();
  int t = blockIdx.x * 256 + threadIdx.x;
  if (t >= B_ * N_ * KNN) return;
  int b = t / (N_ * KNN);
  int rem = t - b * (N_ * KNN);
  int q = rem >> 4;
  const float* px = pos + (size_t)b * 3 * N_;
  int i2 = idx[t];  // idx flat layout == pixel layout
  float qx = px[q], qy = px[N_ + q], qz = px[2 * N_ + q];
  float nx = px[i2], ny = px[N_ + i2], nz = px[2 * N_ + i2];
  float rx = qx - nx, ry = qy - ny, rz = qz - nz;
  int mA = map_idx[b * N_ + q];
  int mB = map_idx[b * N_ + i2];
  const float* aA = P_anchor + ((size_t)b * MANC + mA) * 3;
  const float* aB = P_anchor + ((size_t)b * MANC + mB) * 3;
  float dax = qx - aA[0], day = qy - aA[1], daz = qz - aA[2];
  float dbx = nx - aB[0], dby = ny - aB[1], dbz = nz - aB[2];
  float dA = sqrtf(dax * dax + day * day + daz * daz);
  float dB = sqrtf(dbx * dbx + dby * dby + dbz * dbz);
  float dAB = anchor_dist[((size_t)b * MANC + mA) * MANC + mB];
  float R2 = dA + dAB + dB;
  float feats[6] = {fabsf(rx), fabsf(ry), fabsf(rz), R2, R2, R2};
  float h32[32];
#pragma unroll
  for (int r = 0; r < 32; ++r) {
    float s = 0.f;
#pragma unroll
    for (int c = 0; c < 6; ++c) s += s_lw1[r * 6 + c] * feats[c];
    h32[r] = fmaxf(s, 0.f);
  }
  size_t pix = (size_t)rem;
  size_t bb = (size_t)b * HP * NPIX;
  for (int r = 0; r < HP; ++r) {
    float s = 0.f;
#pragma unroll
    for (int c = 0; c < 32; ++c) s += s_lw2[r * 32 + c] * h32[c];
    h64l[bb + (size_t)r * NPIX + pix] = (_Float16)fmaxf(s, 0.f);
    float p = s_pw1[r * 3] * rx + s_pw1[r * 3 + 1] * ry + s_pw1[r * 3 + 2] * rz;
    p = (p + s_pm[r]) * s_pm[64 + r] + s_pm[128 + r];
    h64p[bb + (size_t)r * NPIX + pix] = (_Float16)fmaxf(p, 0.f);
  }
}

// ---------------------------------------------------------------------------
// pv = pw2 @ h64p + lw3 @ h64l + pb2  (256 x NPIX per batch, K = 64 + 64)
// epilogue: s = query - key[idx] + pv   (both stored f16)
__global__ void gemm_pe(const float* __restrict__ pw2,
                        const float* __restrict__ lw3,
                        const float* __restrict__ pb2,
                        const _Float16* __restrict__ h64p,
                        const _Float16* __restrict__ h64l,
                        const float* __restrict__ query,
                        const float* __restrict__ key,
                        const int* __restrict__ idx, _Float16* __restrict__ pv,
                        _Float16* __restrict__ s_out) {
  int lane = threadIdx.x & 31;
  int half = lane >> 4, row = lane & 15;
  int n0 = blockIdx.x * 16, o0 = blockIdx.y * 16, b = blockIdx.z;
  const _Float16* hp = h64p + (size_t)b * HP * NPIX;
  const _Float16* hl = h64l + (size_t)b * HP * NPIX;
  v8f acc = {};
#pragma unroll
  for (int k0 = 0; k0 < HP; k0 += 32)
    acc = wmma32f16(frag_a_f32(pw2, o0 + row, HP, k0, half),
                    frag_b_f16(hp, k0, (size_t)NPIX, n0 + row, half), acc);
#pragma unroll
  for (int k0 = 0; k0 < HP; k0 += 32)
    acc = wmma32f16(frag_a_f32(lw3, o0 + row, HP, k0, half),
                    frag_b_f16(hl, k0, (size_t)NPIX, n0 + row, half), acc);
  int col = n0 + row;
  int q = col >> 4, j = col & 15;
  int i2 = idx[((size_t)b * N_ + q) * KNN + j];
#pragma unroll
  for (int v = 0; v < 8; ++v) {
    int c = o0 + v + 8 * half;
    float pvv = acc[v] + pb2[c];
    size_t off = ((size_t)b * DIM + c) * NPIX + col;
    pv[off] = (_Float16)pvv;
    float sv = query[((size_t)b * DIM + c) * N_ + q] -
               key[((size_t)b * DIM + c) * N_ + i2] + pvv;
    s_out[off] = (_Float16)sv;
  }
}

// ---------------------------------------------------------------------------
// Fused attention MLP + softmax + aggregation, one block per (q,b), 8 waves.
// Stage 1: a = relu(bn(aw1 @ s)) kept entirely in LDS, stored TRANSPOSED
//          [j][ch] so stage-2 B fragments are contiguous 32B LDS loads.
// Stage 2: logits = aw2 @ a; softmax over the 16 neighbors (intra-half-wave
//          shuffles); agg = sum_j attn * (value + pv).
// ab2 is constant along the softmax axis -> cancels.
__global__ __launch_bounds__(256) void attn_kernel(
    const _Float16* __restrict__ aw1h, const float* __restrict__ ab1,
    const float* __restrict__ ag1, const float* __restrict__ abt1,
    const _Float16* __restrict__ aw2h, const _Float16* __restrict__ s_in,
    const _Float16* __restrict__ pv, const float* __restrict__ value,
    float* __restrict__ agg) {
  __shared__ alignas(32) _Float16 sh_s[DIM * 16];  // [c][j], 8 KB
  __shared__ alignas(32) _Float16 sh_aT[16 * HA];  // [j][ch], 32 KB
  int q = blockIdx.x, b = blockIdx.y;
  int tid = threadIdx.x;
  const _Float16* sb = s_in + (size_t)b * DIM * NPIX + (size_t)q * 16;
  // ---- stage s-tile into LDS: 512 chunks of 16 B ----
#if USE_ASYNC_LDS
  for (int e = tid; e < DIM * 2; e += 256) {
    int c = e >> 1, hr = e & 1;
    __builtin_amdgcn_global_load_async_to_lds_b128(
        (gas_v4i)(sb + (size_t)c * NPIX + hr * 8),
        (las_v4i)(sh_s + c * 16 + hr * 8), 0, 0);
  }
  wait_async_zero();
#else
  for (int e = tid; e < DIM * 2; e += 256) {
    int c = e >> 1, hr = e & 1;
    *(uint4*)(sh_s + c * 16 + hr * 8) =
        *(const uint4*)(sb + (size_t)c * NPIX + hr * 8);
  }
#endif
  __syncthreads();
  int wave = tid >> 5, lane = tid & 31;
  int half = lane >> 4, row = lane & 15;
  const float bnscale = rsqrtf(1.0f + 1e-5f);
  // ---- stage 1: 1024x16 hidden tile -> sh_aT ----
  for (int t = 0; t < 8; ++t) {
    int m0 = wave * 128 + t * 16;
    v8f acc = {};
#pragma unroll
    for (int k0 = 0; k0 < DIM; k0 += 32) {
      v16h a = frag_a_f16(aw1h, m0 + row, DIM, k0, half);
      v16h bfrag;
      const _Float16* p = sh_s + (size_t)(k0 + half * 16) * 16 + row;
#pragma unroll
      for (int j = 0; j < 16; ++j) bfrag[j] = p[j * 16];
      acc = wmma32f16(a, bfrag, acc);
    }
    union { _Float16 h[8]; uint4 u; } pk;
#pragma unroll
    for (int v = 0; v < 8; ++v) {
      int ch = m0 + v + 8 * half;
      float z = (acc[v] + ab1[ch]) * (ag1[ch] * bnscale) + abt1[ch];
      pk.h[v] = (_Float16)fmaxf(z, 0.f);
    }
    // 8 consecutive channels per lane -> one 16 B LDS store (transposed)
    *(uint4*)(sh_aT + (size_t)row * HA + m0 + 8 * half) = pk.u;
  }
  __syncthreads();
  // ---- stage 2: logits + softmax + aggregate ----
  for (int t = 0; t < 2; ++t) {
    int m0 = wave * 32 + t * 16;
    v8f acc = {};
#pragma unroll 4
    for (int k0 = 0; k0 < HA; k0 += 32) {
      v16h a = frag_a_f16(aw2h, m0 + row, HA, k0, half);
      v16h bfrag = *(const v16h*)(sh_aT + (size_t)row * HA + k0 + half * 16);
      acc = wmma32f16(a, bfrag, acc);
    }
#pragma unroll
    for (int v = 0; v < 8; ++v) {
      int c = m0 + v + 8 * half;
      float lg = acc[v];
      float mx = redmax16(lg);
      float e = __expf(lg - mx);
      float sm = redsum16(e);
      float attn = e / sm;
      float ve = value[((size_t)b * DIM + c) * N_ + q] +
                 (float)pv[((size_t)b * DIM + c) * NPIX + q * 16 + row];
      float contrib = redsum16(attn * ve);
      if (row == 0) agg[((size_t)b * DIM + c) * N_ + q] = contrib;
    }
  }
}

// ---------------------------------------------------------------------------
extern "C" void kernel_launch(void* const* d_in, const int* in_sizes, int n_in,
                              void* d_out, int out_size, void* d_ws,
                              size_t ws_size, hipStream_t stream) {
  const float* x           = (const float*)d_in[0];
  const float* pos         = (const float*)d_in[1];
  /* d_in[2] fps_idx: unused by the reference */
  const float* P_anchor    = (const float*)d_in[3];
  const int*   map_idx     = (const int*)d_in[4];
  const float* anchor_dist = (const float*)d_in[5];
  const float* w_start = (const float*)d_in[6];
  const float* b_start = (const float*)d_in[7];
  const float* w_key   = (const float*)d_in[8];
  const float* b_key   = (const float*)d_in[9];
  const float* w_query = (const float*)d_in[10];
  const float* b_query = (const float*)d_in[11];
  const float* w_value = (const float*)d_in[12];
  const float* b_value = (const float*)d_in[13];
  const float* pw1  = (const float*)d_in[14];
  const float* pb1  = (const float*)d_in[15];
  const float* pg1  = (const float*)d_in[16];
  const float* pbt1 = (const float*)d_in[17];
  const float* pw2  = (const float*)d_in[18];
  const float* pb2  = (const float*)d_in[19];
  const float* lw1  = (const float*)d_in[20];
  const float* lw2  = (const float*)d_in[21];
  const float* lw3  = (const float*)d_in[22];
  const float* aw1  = (const float*)d_in[23];
  const float* ab1  = (const float*)d_in[24];
  const float* ag1  = (const float*)d_in[25];
  const float* abt1 = (const float*)d_in[26];
  const float* aw2  = (const float*)d_in[27];
  /* d_in[28] ab2: constant along softmax axis -> cancels */
  const float* w_end = (const float*)d_in[29];
  const float* b_end = (const float*)d_in[30];
  float* y = (float*)d_out;
  (void)in_sizes; (void)n_in; (void)out_size; (void)ws_size;

  // workspace carve-out (~105 MB total)
  char* ws = (char*)d_ws;
  size_t off = 0;
  auto carve = [&](size_t bytes) -> void* {
    void* p = (void*)(ws + off);
    off += (bytes + 255) & ~(size_t)255;
    return p;
  };
  int*      idx  = (int*)     carve((size_t)B_ * N_ * KNN * 4);
  float*    xf   = (float*)   carve((size_t)B_ * DIM * N_ * 4);
  float*    keyb = (float*)   carve((size_t)B_ * DIM * N_ * 4);
  float*    qryb = (float*)   carve((size_t)B_ * DIM * N_ * 4);
  float*    valb = (float*)   carve((size_t)B_ * DIM * N_ * 4);
  float*    aggb = (float*)   carve((size_t)B_ * DIM * N_ * 4);
  _Float16* h64p = (_Float16*)carve((size_t)B_ * HP * NPIX * 2);
  _Float16* h64l = (_Float16*)carve((size_t)B_ * HP * NPIX * 2);
  _Float16* pvb  = (_Float16*)carve((size_t)B_ * DIM * NPIX * 2);
  _Float16* sb   = (_Float16*)carve((size_t)B_ * DIM * NPIX * 2);
  _Float16* aw1h = (_Float16*)carve((size_t)HA * DIM * 2);
  _Float16* aw2h = (_Float16*)carve((size_t)DIM * HA * 2);

  cvt_f32_to_f16<<<256, 256, 0, stream>>>(aw1, aw1h, HA * DIM);
  cvt_f32_to_f16<<<256, 256, 0, stream>>>(aw2, aw2h, DIM * HA);
  knn_kernel<<<(B_ * N_ + 255) / 256, 256, 0, stream>>>(pos, idx);
  gemm_wave<false><<<dim3(N_ / 64, DIM / 128, B_), 256, 0, stream>>>(
      w_start, x, b_start, nullptr, xf, DIM, CIN, N_);
  gemm_wave<false><<<dim3(N_ / 64, DIM / 128, B_), 256, 0, stream>>>(
      w_key, xf, b_key, nullptr, keyb, DIM, DIM, N_);
  gemm_wave<false><<<dim3(N_ / 64, DIM / 128, B_), 256, 0, stream>>>(
      w_query, xf, b_query, nullptr, qryb, DIM, DIM, N_);
  gemm_wave<false><<<dim3(N_ / 64, DIM / 128, B_), 256, 0, stream>>>(
      w_value, xf, b_value, nullptr, valb, DIM, DIM, N_);
  prep_kernel<<<(B_ * N_ * KNN + 255) / 256, 256, 0, stream>>>(
      pos, idx, map_idx, P_anchor, anchor_dist, lw1, lw2, pw1, pb1, pg1, pbt1,
      h64l, h64p);
  gemm_pe<<<dim3(NPIX / 16, DIM / 16, B_), 32, 0, stream>>>(
      pw2, lw3, pb2, h64p, h64l, qryb, keyb, idx, pvb, sb);
  attn_kernel<<<dim3(N_, B_), 256, 0, stream>>>(aw1h, ab1, ag1, abt1, aw2h, sb,
                                                pvb, valb, aggb);
  gemm_wave<true><<<dim3(N_ / 64, CIN / 128, B_), 256, 0, stream>>>(
      w_end, aggb, b_end, x, y, CIN, DIM, N_);
}
